// CantorCrossAttention_84052509983128
// MI455X (gfx1250) — matrix operand order repaired and checked by reading, not compiled, not run on hardware
//
#include <hip/hip_runtime.h>

// ---------------- problem constants (match reference) ----------------
#define DIM_    1024
#define HEADS_  16
#define SEQ_    2048
#define HD_     64
#define B_      2
#define DEPTH_  7
#define LOCALW_ 64
#define SCALE_  0.125f   // 1/sqrt(64)

typedef __bf16 bf16;
typedef __attribute__((ext_vector_type(16))) __bf16 v16bf;
typedef __attribute__((ext_vector_type(8)))  __bf16 v8bf;
typedef __attribute__((ext_vector_type(8)))  float  v8f;
typedef __attribute__((ext_vector_type(4)))  int    v4i;

#define AS1 __attribute__((address_space(1)))
#define AS3 __attribute__((address_space(3)))

// f32 -> bf16 round-to-nearest-even
static __device__ __forceinline__ bf16 f2bf(float x) {
    unsigned u = __builtin_bit_cast(unsigned, x);
    u += 0x7fffu + ((u >> 16) & 1u);
    unsigned short h = (unsigned short)(u >> 16);
    return __builtin_bit_cast(bf16, h);
}

static __device__ __forceinline__ v16bf cat16(v8bf lo, v8bf hi) {
    v16bf r;
#pragma unroll
    for (int i = 0; i < 8; ++i) { r[i] = lo[i]; r[i + 8] = hi[i]; }
    return r;
}

static __device__ __forceinline__ v8bf ld8(const bf16* p) {
    return *reinterpret_cast<const v8bf*>(p);
}

// ---- CDNA5 async global->LDS copy (16B per lane) + asynccnt wait ----
static __device__ __forceinline__ void async_cp16(const bf16* g, bf16* l) {
#if __has_builtin(__builtin_amdgcn_global_load_async_to_lds_b128)
    __builtin_amdgcn_global_load_async_to_lds_b128(
        (AS1 v4i*)g, (AS3 v4i*)l, 0, 0);
#else
    unsigned loff = (unsigned)(unsigned long long)l;   // low 32 bits = LDS offset
    asm volatile("global_load_async_to_lds_b128 %0, %1, off"
                 :: "v"(loff), "v"(g) : "memory");
#endif
}

#if __has_builtin(__builtin_amdgcn_s_wait_asynccnt)
#define WAIT_ASYNC(n) __builtin_amdgcn_s_wait_asynccnt(n)
#else
#define WAIT_ASYNC(n) asm volatile("s_wait_asynccnt %0" :: "n"(n) : "memory")
#endif

// ---------------- stage 0a: Cantor offset table ----------------
__global__ void cantor_mask_kernel(unsigned char* __restrict__ mask) {
    int d = blockIdx.x * blockDim.x + threadIdx.x;
    if (d >= SEQ_) return;
    int x = d;
    int ok = 1;
#pragma unroll
    for (int i = 0; i < DEPTH_; ++i) { ok &= ((x % 3) != 1) ? 1 : 0; x /= 3; }
    ok &= (x == 0) ? 1 : 0;
    mask[d] = (ok || d <= LOCALW_) ? 1 : 0;
}

// ---------------- stage 0b: weight transpose + bf16 convert ----------------
// W [K][N] f32  ->  Wt [N][K] bf16   (makes WMMA B-fragments contiguous)
__global__ void wtrans_kernel(const float* __restrict__ W, bf16* __restrict__ Wt,
                              int K, int N) {
    int idx = blockIdx.x * blockDim.x + threadIdx.x;
    if (idx >= K * N) return;
    int n = idx / K, k = idx % K;
    Wt[idx] = f2bf(W[(size_t)k * N + n]);
}

// ---------------- stage 1/2: input projections (bf16 WMMA, 16x64 tiles) ------
// mode 0: X@W+bias -> outA as q layout [B][H][S][hd]        (N = DIM)
// mode 1: cols <  DIM -> outA  as k layout [B][H][S][hd]    (N = 2*DIM)
//         cols >= DIM -> outVT as v^T layout [B][H][hd][S]
__global__ void proj_wmma_kernel(const float* __restrict__ X,
                                 const bf16* __restrict__ Wt,   // [N][DIM] bf16
                                 const float* __restrict__ bias,
                                 bf16* __restrict__ outA,
                                 bf16* __restrict__ outVT,
                                 int N, int mode) {
    const int lane = threadIdx.x;
    const int hl   = lane >> 4;
    const int n    = lane & 15;
    const int row0 = blockIdx.y * 16;
    const int col0 = blockIdx.x * 64;

    const float* arow = X + (size_t)(row0 + n) * DIM_;   // A row m = lane%16
    v8f acc[4];
    const v8f vzero = {};
#pragma unroll
    for (int t = 0; t < 4; ++t) acc[t] = vzero;

#pragma unroll 1
    for (int k0 = 0; k0 < DIM_; k0 += 32) {
        // A fragment (16x32): elems 0..15 -> K in [k0+hl*16,+16) (contiguous)
        const float* apLo = arow + k0 + hl * 16;
        const float* apHi = arow + k0 + hl * 16 + 8;
        v16bf a;
#pragma unroll
        for (int i = 0; i < 8; ++i) { a[i] = f2bf(apLo[i]); a[i + 8] = f2bf(apHi[i]); }
#pragma unroll
        for (int t = 0; t < 4; ++t) {
            // B fragment: lane column (col0+t*16+n), K = k0 + hl*16 + e (contiguous)
            const bf16* wr = Wt + (size_t)(col0 + t * 16 + n) * DIM_ + k0 + hl * 16;
            const v16bf b = cat16(ld8(wr), ld8(wr + 8));
            acc[t] = __builtin_amdgcn_wmma_f32_16x16x32_bf16(false, a, false, b,
                                                             (short)0, acc[t], false, false);
        }
    }
#pragma unroll
    for (int t = 0; t < 4; ++t) {
        const int col  = col0 + t * 16 + n;
        const float badd = bias[col];
#pragma unroll
        for (int v = 0; v < 8; ++v) {
            const int row = row0 + v + hl * 8;     // C layout: row = v + half*8
            const int bb  = row >> 11;             // / SEQ_
            const int s   = row & (SEQ_ - 1);
            const float val = acc[t][v] + badd;
            if (mode == 0 || col < DIM_) {
                const int c = col;
                const int h = c >> 6, d = c & 63;
                outA[((size_t)(bb * HEADS_ + h) * SEQ_ + s) * HD_ + d] = f2bf(val);
            } else {
                const int c = col - DIM_;
                const int h = c >> 6, d = c & 63;
                outVT[((size_t)(bb * HEADS_ + h) * HD_ + d) * SEQ_ + s] = f2bf(val);
            }
        }
    }
}

// ---------------- stage 3: flash attention, async double-buffered K/V -------
__global__ void attn_wmma_kernel(const bf16* __restrict__ q,
                                 const bf16* __restrict__ k,
                                 const bf16* __restrict__ vT,
                                 const unsigned char* __restrict__ maskTab,
                                 bf16* __restrict__ oOut) {
    __shared__ unsigned char smask[SEQ_];
    __shared__ alignas(16) bf16 sP[16 * 32];          // P tile (transpose staging)
    __shared__ alignas(16) bf16 skbuf[2][32 * HD_];   // K chunk: 32 keys x 64
    __shared__ alignas(16) bf16 svbuf[2][HD_ * 32];   // V^T chunk: 64 dims x 32 keys

    const int lane = threadIdx.x;
    const int hl   = lane >> 4;
    const int n    = lane & 15;
    const int bh   = blockIdx.y;                      // b*HEADS + h
    const int q0   = blockIdx.x * 16;

    for (int i = lane; i < SEQ_; i += 32) smask[i] = maskTab[i];
    __syncthreads();

    const bf16* qh = q  + (size_t)bh * SEQ_ * HD_;
    const bf16* kh = k  + (size_t)bh * SEQ_ * HD_;
    const bf16* vh = vT + (size_t)bh * HD_ * SEQ_;

    // Q A-fragments for hd chunks [0,32) and [32,64), loaded once
    const bf16* qrow = qh + (size_t)(q0 + n) * HD_;
    const v16bf a0 = cat16(ld8(qrow + hl * 16),      ld8(qrow + hl * 16 + 8));
    const v16bf a1 = cat16(ld8(qrow + 32 + hl * 16), ld8(qrow + 32 + hl * 16 + 8));

    float mrow[8], lrow[8];
    v8f accO[4];
    const v8f vzero = {};
#pragma unroll
    for (int v = 0; v < 8; ++v) { mrow[v] = -3.0e38f; lrow[v] = 0.f; }
#pragma unroll
    for (int t = 0; t < 4; ++t) accO[t] = vzero;

    // issue async copies of one 32-key chunk (16 instructions, 16B/lane each)
    auto issue_chunk = [&](int j0, int buf) {
        // K: 32 rows x 128B; lane owns row j0+lane
        const bf16* gk = kh + (size_t)(j0 + lane) * HD_;
        bf16* lk = &skbuf[buf][lane * HD_];
#pragma unroll
        for (int t = 0; t < 8; ++t) async_cp16(gk + t * 8, lk + t * 8);
        // V^T: 64 rows x 64B; 8 rows per instruction, 4 lanes per row
        const int vr  = lane >> 2;
        const int voc = (lane & 3) * 8;
#pragma unroll
        for (int t = 0; t < 8; ++t) {
            const int r = t * 8 + vr;
            async_cp16(vh + (size_t)r * SEQ_ + j0 + voc, &svbuf[buf][r * 32 + voc]);
        }
    };

    issue_chunk(0, 0);

    const int NCHUNK = SEQ_ / 32;   // 64
#pragma unroll 1
    for (int i = 0; i < NCHUNK; ++i) {
        const int j0  = i * 32;
        const int buf = i & 1;
        if (i + 1 < NCHUNK) {
            issue_chunk(j0 + 32, buf ^ 1);
            WAIT_ASYNC(16);          // chunk i's 16 copies complete (in-order)
        } else {
            WAIT_ASYNC(0);
        }

        // ---- scores: two 16x16 tiles over 32 keys (B fragments from LDS) ----
        v8f s0 = {}, s1 = {};
        {
            const bf16* kr0 = &skbuf[buf][(size_t)n * HD_];
            const v16bf b00 = cat16(ld8(kr0 + hl * 16),      ld8(kr0 + hl * 16 + 8));
            const v16bf b01 = cat16(ld8(kr0 + 32 + hl * 16), ld8(kr0 + 32 + hl * 16 + 8));
            s0 = __builtin_amdgcn_wmma_f32_16x16x32_bf16(false, a0, false, b00, (short)0, s0, false, false);
            s0 = __builtin_amdgcn_wmma_f32_16x16x32_bf16(false, a1, false, b01, (short)0, s0, false, false);
            const bf16* kr1 = &skbuf[buf][(size_t)(16 + n) * HD_];
            const v16bf b10 = cat16(ld8(kr1 + hl * 16),      ld8(kr1 + hl * 16 + 8));
            const v16bf b11 = cat16(ld8(kr1 + 32 + hl * 16), ld8(kr1 + 32 + hl * 16 + 8));
            s1 = __builtin_amdgcn_wmma_f32_16x16x32_bf16(false, a0, false, b10, (short)0, s1, false, false);
            s1 = __builtin_amdgcn_wmma_f32_16x16x32_bf16(false, a1, false, b11, (short)0, s1, false, false);
        }

        // ---- scale + Cantor mask (C layout: row = v + hl*8, col = n) ----
#pragma unroll
        for (int v = 0; v < 8; ++v) {
            const int qi = q0 + v + hl * 8;
            int d0 = qi - (j0 + n);      d0 = d0 < 0 ? -d0 : d0;
            int d1 = qi - (j0 + 16 + n); d1 = d1 < 0 ? -d1 : d1;
            const float x0 = s0[v] * SCALE_;
            const float x1 = s1[v] * SCALE_;
            s0[v] = smask[d0] ? x0 : -3.0e38f;
            s1[v] = smask[d1] ? x1 : -3.0e38f;
        }

        // ---- online softmax update (reductions within half-wave 16-groups) ----
        float alpha[8];
#pragma unroll
        for (int v = 0; v < 8; ++v) {
            float x = fmaxf(s0[v], s1[v]);
#pragma unroll
            for (int m = 1; m < 16; m <<= 1) x = fmaxf(x, __shfl_xor(x, m, 32));
            const float mnew = fmaxf(mrow[v], x);
            alpha[v] = __expf(mrow[v] - mnew);
            mrow[v]  = mnew;
            const float p0 = __expf(s0[v] - mnew);
            const float p1 = __expf(s1[v] - mnew);
            s0[v] = p0; s1[v] = p1;
            float sum = p0 + p1;
#pragma unroll
            for (int m = 1; m < 16; m <<= 1) sum += __shfl_xor(sum, m, 32);
            lrow[v] = lrow[v] * alpha[v] + sum;
        }
#pragma unroll
        for (int t = 0; t < 4; ++t)
#pragma unroll
            for (int v = 0; v < 8; ++v) accO[t][v] *= alpha[v];

        // ---- transpose P (C layout -> A layout) through LDS ----
        __syncthreads();
#pragma unroll
        for (int v = 0; v < 8; ++v) {
            sP[(v + hl * 8) * 32 + n]      = f2bf(s0[v]);
            sP[(v + hl * 8) * 32 + 16 + n] = f2bf(s1[v]);
        }
        __syncthreads();
        const bf16* prow = &sP[n * 32];
        const v16bf pa = cat16(ld8(prow + hl * 16), ld8(prow + hl * 16 + 8));

        // ---- accO += P @ V  (V^T chunk in LDS gives contiguous B fragments) ----
#pragma unroll
        for (int t = 0; t < 4; ++t) {
            const bf16* vrow = &svbuf[buf][(size_t)(t * 16 + n) * 32];
            const v16bf bv = cat16(ld8(vrow + hl * 16), ld8(vrow + hl * 16 + 8));
            accO[t] = __builtin_amdgcn_wmma_f32_16x16x32_bf16(false, pa, false, bv,
                                                              (short)0, accO[t], false, false);
        }
    }

    // ---- normalize and store O as [B*S][DIM] bf16 ----
    const int bb = bh / HEADS_;
    const int h  = bh % HEADS_;
#pragma unroll
    for (int t = 0; t < 4; ++t)
#pragma unroll
        for (int v = 0; v < 8; ++v) {
            const int srow = q0 + v + hl * 8;
            const float val = accO[t][v] / lrow[v];
            oOut[((size_t)(bb * SEQ_ + srow)) * DIM_ + h * HD_ + t * 16 + n] = f2bf(val);
        }
}

// ---------------- stage 4: output projection (bf16 A, 16x64 tiles) ----------
__global__ void oproj_wmma_kernel(const bf16* __restrict__ O,
                                  const bf16* __restrict__ WoT,  // [DIM][DIM] bf16
                                  const float* __restrict__ bo,
                                  float* __restrict__ out) {
    const int lane = threadIdx.x;
    const int hl   = lane >> 4;
    const int n    = lane & 15;
    const int row0 = blockIdx.y * 16;
    const int col0 = blockIdx.x * 64;

    const bf16* arow = O + (size_t)(row0 + n) * DIM_;
    v8f acc[4];
    const v8f vzero = {};
#pragma unroll
    for (int t = 0; t < 4; ++t) acc[t] = vzero;

#pragma unroll 1
    for (int k0 = 0; k0 < DIM_; k0 += 32) {
        const v16bf a = cat16(ld8(arow + k0 + hl * 16), ld8(arow + k0 + hl * 16 + 8));
#pragma unroll
        for (int t = 0; t < 4; ++t) {
            const bf16* wr = WoT + (size_t)(col0 + t * 16 + n) * DIM_ + k0 + hl * 16;
            const v16bf b = cat16(ld8(wr), ld8(wr + 8));
            acc[t] = __builtin_amdgcn_wmma_f32_16x16x32_bf16(false, a, false, b,
                                                             (short)0, acc[t], false, false);
        }
    }
#pragma unroll
    for (int t = 0; t < 4; ++t) {
        const int col = col0 + t * 16 + n;
        const float badd = bo[col];
#pragma unroll
        for (int v = 0; v < 8; ++v) {
            const int row = row0 + v + hl * 8;
            out[(size_t)row * DIM_ + col] = acc[t][v] + badd;
        }
    }
}

// ---------------- launcher ----------------
extern "C" void kernel_launch(void* const* d_in, const int* in_sizes, int n_in,
                              void* d_out, int out_size, void* d_ws, size_t ws_size,
                              hipStream_t stream) {
    const float* query  = (const float*)d_in[0];
    const float* keyval = (const float*)d_in[1];
    const float* Wq     = (const float*)d_in[2];
    const float* bq     = (const float*)d_in[3];
    const float* Wkv    = (const float*)d_in[4];
    const float* bkv    = (const float*)d_in[5];
    const float* Wo     = (const float*)d_in[6];
    const float* bo     = (const float*)d_in[7];
    float* out = (float*)d_out;

    // workspace (~40 MB): q, k, v^T, o (bf16), transposed bf16 weights, mask
    char* ws = (char*)d_ws;
    const size_t nQ = (size_t)B_ * HEADS_ * SEQ_ * HD_;      // 4,194,304 elems
    bf16* qbf  = (bf16*)ws;                    ws += nQ * sizeof(bf16);
    bf16* kbf  = (bf16*)ws;                    ws += nQ * sizeof(bf16);
    bf16* vtbf = (bf16*)ws;                    ws += nQ * sizeof(bf16);
    bf16* obf  = (bf16*)ws;                    ws += (size_t)B_ * SEQ_ * DIM_ * sizeof(bf16);
    bf16* WqT  = (bf16*)ws;                    ws += (size_t)DIM_ * DIM_ * sizeof(bf16);
    bf16* WkvT = (bf16*)ws;                    ws += (size_t)DIM_ * 2 * DIM_ * sizeof(bf16);
    bf16* WoT  = (bf16*)ws;                    ws += (size_t)DIM_ * DIM_ * sizeof(bf16);
    unsigned char* mask = (unsigned char*)ws;

    cantor_mask_kernel<<<dim3(SEQ_ / 256), dim3(256), 0, stream>>>(mask);

    wtrans_kernel<<<dim3((DIM_ * DIM_) / 256),     dim3(256), 0, stream>>>(Wq,  WqT,  DIM_, DIM_);
    wtrans_kernel<<<dim3((DIM_ * 2 * DIM_) / 256), dim3(256), 0, stream>>>(Wkv, WkvT, DIM_, 2 * DIM_);
    wtrans_kernel<<<dim3((DIM_ * DIM_) / 256),     dim3(256), 0, stream>>>(Wo,  WoT,  DIM_, DIM_);

    proj_wmma_kernel<<<dim3(DIM_ / 64, (B_ * SEQ_) / 16), dim3(32), 0, stream>>>(
        query, WqT, bq, qbf, nullptr, DIM_, 0);

    proj_wmma_kernel<<<dim3((2 * DIM_) / 64, (B_ * SEQ_) / 16), dim3(32), 0, stream>>>(
        keyval, WkvT, bkv, kbf, vtbf, 2 * DIM_, 1);

    attn_wmma_kernel<<<dim3(SEQ_ / 16, B_ * HEADS_), dim3(32), 0, stream>>>(
        qbf, kbf, vtbf, mask, obf);

    oproj_wmma_kernel<<<dim3(DIM_ / 64, (B_ * SEQ_) / 16), dim3(32), 0, stream>>>(
        obf, WoT, bo, out);
}
